// ConvNorm_43791486550487
// MI455X (gfx1250) — matrix operand order repaired and checked by reading, not compiled, not run on hardware
//
#include <hip/hip_runtime.h>

// ---------------------------------------------------------------------------
// DoReFa-quantized 3x3 conv + BN + PACT as implicit GEMM on CDNA5 WMMA f16.
// B=32, Cin=Cout=256, H=W=56. GEMM: M=100352, N=256, K=2304.
// k ordering: k = (kh*3+kw)*256 + c -> every 64-wide K-slice has fixed (kh,kw).
// x pre-transposed to [b][hw][c] f16 -> all A global traffic is 16B loads.
// Weights pre-quantized+packed: BqT[n][k]; B tiles staged by the Tensor Data
// Mover (tensor_load_to_lds, TENSORcnt) with TDM LDS padding matching the
// 144B-padded LDS rows. 128x128 tile/block, 8 waves, K-step 64 => 16 WMMAs
// per wave per macro-step, double-buffered LDS, software-pipelined.
// Workspace: BqT 1.18MB + BN tables + xt 51.4MB (~52.6 MB total).
// ---------------------------------------------------------------------------

typedef __attribute__((ext_vector_type(16))) _Float16 v16h;
typedef __attribute__((ext_vector_type(8)))  _Float16 half8;
typedef __attribute__((ext_vector_type(8)))  float    v8f;
typedef __attribute__((ext_vector_type(4)))  unsigned v4u;
typedef __attribute__((ext_vector_type(8)))  int      v8i;
typedef __attribute__((ext_vector_type(4)))  int      v4i;

#define CIN   256
#define COUT  256
#define HDIM  56
#define WDIM  56
#define HWSZ  3136          // 56*56
#define KTOT  2304          // 256*9
#define TM    128
#define TN    128
#define KS    64            // K per LDS buffer
#define NKBIG 36            // 2304/64 macro-steps (even)
#define LDSP  72            // padded LDS row stride in halves (144B)
#define BUF_HALVES (TM * LDSP)       // 9216 halves
#define BUF_BYTES  (BUF_HALVES * 2)  // 18432 B per buffer
#define A0_OFF 0
#define A1_OFF (1 * BUF_BYTES)
#define B0_OFF (2 * BUF_BYTES)
#define B1_OFF (3 * BUF_BYTES)

__device__ __forceinline__ v16h combine16(half8 lo, half8 hi) {
    v16h r;
#pragma unroll
    for (int i = 0; i < 8; ++i) { r[i] = lo[i]; r[i + 8] = hi[i]; }
    return r;
}

// ---------------- preprocessing ----------------

__global__ void init_ws_kernel(unsigned* wmax) {
    if (threadIdx.x == 0 && blockIdx.x == 0) *wmax = 0u;
}

__global__ void wmax_kernel(const float* __restrict__ w, unsigned* wmax, int n) {
    __shared__ float red[256];
    float m = 0.0f;
    for (int i = blockIdx.x * blockDim.x + threadIdx.x; i < n; i += gridDim.x * blockDim.x)
        m = fmaxf(m, fabsf(tanhf(w[i])));
    red[threadIdx.x] = m;
    __syncthreads();
    for (int s = 128; s > 0; s >>= 1) {
        if ((int)threadIdx.x < s) red[threadIdx.x] = fmaxf(red[threadIdx.x], red[threadIdx.x + s]);
        __syncthreads();
    }
    if (threadIdx.x == 0) atomicMax(wmax, __float_as_uint(red[0]));  // floats >= 0: bit order ok
}

// DoReFa 4-bit quantize + pack transposed: BqT[o*KTOT + (kh*3+kw)*256 + c]
__global__ void pack_kernel(const float* __restrict__ w, const unsigned* __restrict__ wmax,
                            _Float16* __restrict__ BqT, int n) {
    int i = blockIdx.x * blockDim.x + threadIdx.x;
    if (i >= n) return;
    float mx = __uint_as_float(*wmax);
    float inv2m = 0.5f / mx;
    int kw_ = i % 3;
    int t1  = i / 3;
    int kh_ = t1 % 3;
    int t2  = t1 / 3;
    int c   = t2 % CIN;
    int o   = t2 / CIN;
    float t  = tanhf(w[i]) * inv2m + 0.5f;
    float tq = rintf(t * 15.0f) * (1.0f / 15.0f);
    BqT[(size_t)o * KTOT + (kh_ * 3 + kw_) * 256 + c] = (_Float16)(2.0f * tq - 1.0f);
}

__global__ void bn_kernel(const float* __restrict__ gamma, const float* __restrict__ beta,
                          const float* __restrict__ mean,  const float* __restrict__ var,
                          float* __restrict__ invs, float* __restrict__ shifts) {
    int i = threadIdx.x;
    float iv = gamma[i] * rsqrtf(var[i] + 1e-5f);
    invs[i] = iv;
    shifts[i] = beta[i] - mean[i] * iv;
}

// NCHW f32 -> [b][hw][c] f16 tiled transpose (64x64 tiles through LDS)
__global__ __launch_bounds__(256)
void xpose_kernel(const float* __restrict__ x, _Float16* __restrict__ xt) {
    __shared__ _Float16 tile[64][72];
    const int c0  = blockIdx.x * 64;
    const int hw0 = blockIdx.y * 64;
    const int b   = blockIdx.z;
    const int tx  = threadIdx.x & 63;
    const int ty  = threadIdx.x >> 6;   // 0..3
#pragma unroll
    for (int i = 0; i < 16; ++i) {
        int row = ty + i * 4;           // c_local
        tile[row][tx] = (_Float16)x[((b * CIN + c0 + row) * HWSZ) + hw0 + tx];
    }
    __syncthreads();
#pragma unroll
    for (int i = 0; i < 16; ++i) {
        int row = ty + i * 4;           // hw_local
        xt[((size_t)(b * HWSZ + hw0 + row)) * CIN + c0 + tx] = tile[tx][row];
    }
}

// ---------------- TDM descriptor for one B tile (128 rows x 64 f16) ----------------
// D# per CDNA5 ISA. LDS pad: after every 32 DWORDs
// (one 128B row) insert 4 DWORDs (16B) -> 144B row stride == LDSP.
__device__ __forceinline__ void issue_b_tdm(const _Float16* BqT, int n0, int s, unsigned lds_off) {
    const int khkw = s >> 2;
    const int kcol = khkw * 256 + (s & 3) * 64;
    unsigned long long ga = (unsigned long long)(uintptr_t)(BqT + (size_t)n0 * KTOT + kcol);
    v4u g0;
    g0[0] = 1u;                                   // count=1 (valid user descriptor)
    g0[1] = lds_off;                              // lds_addr [63:32]
    g0[2] = (unsigned)(ga & 0xffffffffu);         // global_addr lo
    g0[3] = (unsigned)(ga >> 32) | (2u << 30);    // global_addr hi | type=2 ("image")
    v8i g1;
    g1[0] = (int)((1u << 16)                      // data_size = 2 bytes
                | (1u << 20)                      // pad_enable
                | (4u << 22)                      // pad_interval: 32 DWORDs
                | (3u << 25));                    // pad_amount: 4 DWORDs
    g1[1] = (int)(64u << 16);                     // tensor_dim0[15:0]=64 (bits 63:48)
    g1[2] = (int)(128u << 16);                    // tensor_dim0 hi=0 | tensor_dim1[15:0]=128
    g1[3] = (int)(64u << 16);                     // tensor_dim1 hi=0 | tile_dim0=64
    g1[4] = (int)(128u);                          // tile_dim1=128 | tile_dim2=0
    g1[5] = (int)(KTOT);                          // tensor_dim0_stride = 2304 elements
    g1[6] = 0;                                    // stride0 hi | tensor_dim1_stride lo (unused, 2D)
    g1[7] = 0;
    v4i z4 = {};
    v8i z8 = {};
    // 6-arg form (this toolchain): (g0, g1, g2, g3, g4, cpol)
    __builtin_amdgcn_tensor_load_to_lds(g0, g1, z4, z4, z8, 0);
}

// ---------------- main implicit-GEMM WMMA kernel ----------------

__global__ __launch_bounds__(256)
void conv_wmma_kernel(const _Float16* __restrict__ xt, const _Float16* __restrict__ BqT,
                      const float* __restrict__ invs, const float* __restrict__ shifts,
                      const float* __restrict__ alpha_p, float* __restrict__ out) {
    __shared__ __align__(16) unsigned char smemRaw[4 * BUF_BYTES];  // 72 KB
    _Float16* A0 = (_Float16*)(smemRaw + A0_OFF);
    _Float16* A1 = (_Float16*)(smemRaw + A1_OFF);
    _Float16* B0 = (_Float16*)(smemRaw + B0_OFF);
    _Float16* B1 = (_Float16*)(smemRaw + B1_OFF);

    const int tid  = threadIdx.x;
    const int lane = tid & 31;
    const int wave = tid >> 5;                // 0..7
    const int n0   = blockIdx.x * TN;         // {0,128}
    const int m0   = blockIdx.y * TM;         // 784 tiles, exact
    const int wm   = (wave & 3) * 32;         // wave: 32(M) x 64(N)
    const int wn   = (wave >> 2) * 64;

    // cooperative A loaders: 4 chunks/thread; chunk c: row r0 + 32c, 8 halves at koff
    const int r0   = tid >> 3;                // 0..31
    const int koff = (tid & 7) * 8;           // 0..56
    int arow[4], pb[4], yy0[4], xx0[4];
#pragma unroll
    for (int c = 0; c < 4; ++c) {
        arow[c] = r0 + 32 * c;
        int mg   = m0 + arow[c];
        int bidx = mg / HWSZ;
        int hw_  = mg % HWSZ;
        pb[c]  = bidx * HWSZ;
        yy0[c] = hw_ / WDIM;
        xx0[c] = hw_ % WDIM;
    }

    v8f acc[2][4];
#pragma unroll
    for (int i = 0; i < 2; ++i)
#pragma unroll
        for (int j = 0; j < 4; ++j) { v8f z = {}; acc[i][j] = z; }

    half8 aR[4];

    auto load_stepA = [&](int s) {           // s: macro-step, 64 K-values
        const int khkw = s >> 2;             // uniform; 4 macro-steps per (kh,kw)
        const int kh = khkw / 3, kw = khkw - kh * 3;
        const int cb = (s & 3) * 64 + koff;  // channel offset within 256
#pragma unroll
        for (int c = 0; c < 4; ++c) {
            int y  = yy0[c] + kh - 1;
            int x2 = xx0[c] + kw - 1;
            bool inb = ((unsigned)y < HDIM) & ((unsigned)x2 < WDIM);
            int idx = inb ? ((pb[c] + y * WDIM + x2) * CIN + cb) : 0;  // clamp -> safe load
            half8 v = *(const half8*)&xt[idx];
            half8 z = {};
            aR[c] = inb ? v : z;
        }
        __builtin_prefetch(&xt[(pb[0] + yy0[0] * WDIM + xx0[0]) * CIN + cb], 0, 1); // global_prefetch_b8
    };

    auto store_stepA = [&](_Float16* A) {
#pragma unroll
        for (int c = 0; c < 4; ++c)
            *(half8*)&A[arow[c] * LDSP + koff] = aR[c];
    };

    auto compute_step = [&](const _Float16* A, const _Float16* B) {
        const int l15 = lane & 15;
        const int hi  = lane >> 4;
        const int ak  = hi * 8;    // A frag (ISA 7.12.2): lanes 0-15 K{0..7,16..23}, 16-31 K{8..15,24..31}
        const int bk  = hi * 8;    // B frag: lanes 0-15 K{0..7,16..23}, 16-31 K{8..15,24..31}
#pragma unroll
        for (int t = 0; t < 2; ++t) {          // two 32-K sub-steps per buffer
            const int kb = t * 32;
            v16h a[2];
#pragma unroll
            for (int i = 0; i < 2; ++i) {
                const int ra = (wm + i * 16 + l15) * LDSP + kb + ak;
                a[i] = combine16(*(const half8*)&A[ra], *(const half8*)&A[ra + 16]);
            }
#pragma unroll
            for (int j = 0; j < 4; ++j) {
                const int rb = (wn + j * 16 + l15) * LDSP + kb + bk;
                v16h b = combine16(*(const half8*)&B[rb], *(const half8*)&B[rb + 16]);
                acc[0][j] = __builtin_amdgcn_wmma_f32_16x16x32_f16(false, a[0], false, b,
                                                                   (short)0, acc[0][j], false, false);
                acc[1][j] = __builtin_amdgcn_wmma_f32_16x16x32_f16(false, a[1], false, b,
                                                                   (short)0, acc[1][j], false, false);
            }
        }
    };

    // ---- software-pipelined, double-buffered K loop (NKBIG even) ----
    // Wave 0 drives the TDM for B tiles; everyone stages A; the barrier that
    // publishes a buffer is preceded (in wave 0) by s_wait_tensorcnt 0.
    if (wave == 0) issue_b_tdm(BqT, n0, 0, B0_OFF);
    load_stepA(0);
    store_stepA(A0);
    if (wave == 0) __builtin_amdgcn_s_wait_tensorcnt(0);
    __syncthreads();

    for (int s = 0; s < NKBIG; s += 2) {
        if (wave == 0) issue_b_tdm(BqT, n0, s + 1, B1_OFF);   // overlaps compute below
        load_stepA(s + 1);                  // global -> regs for next macro-step
        compute_step(A0, B0);               // 16 WMMAs/wave from current buffers
        store_stepA(A1);
        if (wave == 0) __builtin_amdgcn_s_wait_tensorcnt(0);
        __syncthreads();

        if (s + 2 < NKBIG) {
            if (wave == 0) issue_b_tdm(BqT, n0, s + 2, B0_OFF);
            load_stepA(s + 2);
        }
        compute_step(A1, B1);
        if (s + 2 < NKBIG) store_stepA(A0);
        if (wave == 0) __builtin_amdgcn_s_wait_tensorcnt(0);
        __syncthreads();
    }

    // ---- epilogue: BN + PACT, LDS-transposed for coalesced stores ----
    // f32 C/D layout: VGPR r -> M = r + (lane>=16 ? 8 : 0); N = lane%16
    float* ep = (float*)smemRaw + wave * (16 * 17);   // wave-private 16x16(+pad) patch
    const float alpha = alpha_p[0];
    const float qs = 15.0f / alpha;
    const float dq = alpha / 15.0f;
    const int l15 = lane & 15;
    const int hi  = lane >> 4;
#pragma unroll
    for (int i = 0; i < 2; ++i) {
        const int mg2 = m0 + wm + i * 16 + l15;   // 16-row subtile never crosses batch bound
        const int bb  = mg2 / HWSZ;
        const int hw2 = mg2 % HWSZ;
#pragma unroll
        for (int j = 0; j < 4; ++j) {
            const int obase = n0 + wn + j * 16;
#pragma unroll
            for (int r = 0; r < 8; ++r)                       // stage [m_local][n_local]
                ep[(r + hi * 8) * 17 + l15] = acc[i][j][r];
            // read transposed: lane -> m_local = l15; covers n_local = hi*8 + rr
#pragma unroll
            for (int rr = 0; rr < 8; ++rr) {
                const int o = obase + hi * 8 + rr;
                float v = ep[l15 * 17 + hi * 8 + rr] * invs[o] + shifts[o];
                v = fminf(fmaxf(v, 0.0f), alpha);
                out[(size_t)(bb * COUT + o) * HWSZ + hw2] = rintf(v * qs) * dq;
            }
        }
    }
}

// ---------------- launcher ----------------

extern "C" void kernel_launch(void* const* d_in, const int* in_sizes, int n_in,
                              void* d_out, int out_size, void* d_ws, size_t ws_size,
                              hipStream_t stream) {
    (void)in_sizes; (void)n_in; (void)out_size; (void)ws_size;

    const float* x      = (const float*)d_in[0];
    const float* weight = (const float*)d_in[1];
    const float* gamma  = (const float*)d_in[2];
    const float* beta   = (const float*)d_in[3];
    const float* rmean  = (const float*)d_in[4];
    const float* rvar   = (const float*)d_in[5];
    const float* alpha  = (const float*)d_in[6];
    float* out = (float*)d_out;

    // workspace layout (~52.6 MB):
    char* ws = (char*)d_ws;
    _Float16* BqT   = (_Float16*)ws;                       // 1 179 648 B
    float*    invs  = (float*)(ws + 1179648);              // 1 KB
    float*    shfts = invs + COUT;                         // 1 KB
    unsigned* wmax  = (unsigned*)(shfts + COUT);           // 4 B
    _Float16* xt    = (_Float16*)(ws + 1182720);           // 51 380 224 B, 16B aligned

    const int NW = COUT * CIN * 9;  // 589 824

    init_ws_kernel<<<1, 1, 0, stream>>>(wmax);
    wmax_kernel<<<64, 256, 0, stream>>>(weight, wmax, NW);
    pack_kernel<<<(NW + 255) / 256, 256, 0, stream>>>(weight, wmax, BqT, NW);
    bn_kernel<<<1, 256, 0, stream>>>(gamma, beta, rmean, rvar, invs, shfts);

    dim3 tgrid(CIN / 64, HWSZ / 64, 32);   // (4, 49, 32)
    xpose_kernel<<<tgrid, 256, 0, stream>>>(x, xt);

    dim3 grid(COUT / TN, (32 * HWSZ) / TM);   // (2, 784)
    conv_wmma_kernel<<<grid, 256, 0, stream>>>(xt, BqT, invs, shfts, alpha, out);
}
